// MemoryGaussianMoELayer_48893907698290
// MI455X (gfx1250) — matrix operand order repaired and checked by reading, not compiled, not run on hardware
//
#include <hip/hip_runtime.h>
#include <hip/hip_bf16.h>
#include <math.h>

// ---------------------------------------------------------------------------
// MoE (Gaussian-routed, top-2 of 8 experts), fused for MI455X (gfx1250).
//   - routing_kernel : per-token Gaussian log-probs -> softmax -> top-2 gates
//   - moe_fused      : per 32-token block, loops experts; GEMM1+GELU+GEMM2
//                      fused in LDS/registers with v_wmma_f32_16x16x32_bf16.
//                      W1 slabs arrive via TDM (tensor_load_to_lds, double-
//                      buffered, s_wait_tensorcnt); W2 slabs via global b128.
// Gate folded into H rows so expert outputs accumulate in registers.
// ---------------------------------------------------------------------------

#define D_IN   1024
#define D_HID  4096
#define D_OUT  1024
#define NEXP   8
#define TOPK   2
#define BM     32      // tokens per workgroup
#define CHUNK  64      // hidden columns per outer chunk
#define KSLAB  128     // K depth staged per TDM slab (4 WMMA k-steps)
#define NSLAB  (D_IN / KSLAB)

typedef __attribute__((ext_vector_type(16))) __bf16 v16bf;
typedef __attribute__((ext_vector_type(8)))  float  v8f;
typedef unsigned int v4u __attribute__((ext_vector_type(4)));
typedef unsigned int v8u __attribute__((ext_vector_type(8)));

union FragBF { v16bf v; uint4 q[2]; };

__device__ __forceinline__ unsigned short f2bf(float f) {
  unsigned int u = __float_as_uint(f);
  u = (u + 0x7FFFu + ((u >> 16) & 1u)) >> 16;   // round-to-nearest-even
  return (unsigned short)u;
}

// Issue a TDM load of a [KSLAB x CHUNK] fp32 tile (row stride D_HID floats)
// into LDS at byte offset ldsOff. One instruction per issuing wave.
__device__ __forceinline__ void tdm_load_w1(const float* gptr, unsigned ldsOff) {
  unsigned long long ga = (unsigned long long)(size_t)gptr;
  v4u g0;
  g0.x = 1u;                                           // count=1, user mode
  g0.y = ldsOff;                                       // LDS byte address
  g0.z = (unsigned)ga;                                 // global addr [31:0]
  g0.w = (unsigned)((ga >> 32) & 0x01FFFFFFu)          // global addr [56:32]
         | 0x80000000u;                                // type = 2 ("image")
  v8u g1;
  g1.s0 = 0x00020000u;          // wg_mask=0, data_size=2 (4B), no pad/iter
  g1.s1 = 0u;                   // atomic_barrier=0 ; tensor_dim0 lo16 = 0
  g1.s2 = 0x00004000u;          // tensor_dim0 hi16 (=1<<30) ; tensor_dim1 lo16
  g1.s3 = 0x00404000u;          // tensor_dim1 hi16 (=1<<30) ; tile_dim0 = 64
  g1.s4 = (unsigned)KSLAB;      // tile_dim1 = 128 ; tile_dim2 = 0
  g1.s5 = (unsigned)D_HID;      // tensor_dim0_stride lo32 = 4096
  g1.s6 = 0u;                   // stride hi16 ; dim1_stride lo16
  g1.s7 = 0u;                   // dim1_stride hi32
  asm volatile("tensor_load_to_lds %0, %1" :: "s"(g0), "s"(g1) : "memory");
}

// ---------------------------------------------------------------------------
// Routing: one wave32 per token.
// ---------------------------------------------------------------------------
__global__ __launch_bounds__(256)
void routing_kernel(const float* __restrict__ x,
                    const float* __restrict__ mus,
                    const float* __restrict__ ls,
                    float* __restrict__ log_probs_out,
                    float* __restrict__ weights_out,
                    int*   __restrict__ top_idx_out,
                    float* __restrict__ gw,
                    int Ntok) {
  int w = threadIdx.x >> 5, lane = threadIdx.x & 31;
  int t = blockIdx.x * 8 + w;
  if (t >= Ntok) return;

  float acc[NEXP], sls[NEXP];
#pragma unroll
  for (int e = 0; e < NEXP; ++e) { acc[e] = 0.f; sls[e] = 0.f; }

  const float* xp = x + (size_t)t * D_IN;
  for (int d = lane; d < D_IN; d += 32) {
    float xv = xp[d];
#pragma unroll
    for (int e = 0; e < NEXP; ++e) {
      float l    = ls[e * D_IN + d];
      float diff = (xv - mus[e * D_IN + d]) * __expf(-l);
      acc[e] += diff * diff;
      sls[e] += l;
    }
  }
#pragma unroll
  for (int e = 0; e < NEXP; ++e) {
    for (int off = 16; off > 0; off >>= 1) {
      acc[e] += __shfl_xor(acc[e], off, 32);
      sls[e] += __shfl_xor(sls[e], off, 32);
    }
  }
  if (lane == 0) {
    float lp[NEXP], wt[NEXP];
    float mx = -3.4e38f;
#pragma unroll
    for (int e = 0; e < NEXP; ++e) {
      lp[e] = -0.5f * acc[e] - sls[e];
      mx = fmaxf(mx, lp[e]);
    }
    float sum = 0.f;
#pragma unroll
    for (int e = 0; e < NEXP; ++e) { wt[e] = __expf(lp[e] - mx); sum += wt[e]; }
    float inv = 1.f / sum;
#pragma unroll
    for (int e = 0; e < NEXP; ++e) wt[e] *= inv;

    int i0 = 0; float m0v = wt[0];
#pragma unroll
    for (int e = 1; e < NEXP; ++e) if (wt[e] > m0v) { m0v = wt[e]; i0 = e; }
    int i1 = (i0 == 0) ? 1 : 0; float m1v = wt[i1];
#pragma unroll
    for (int e = 0; e < NEXP; ++e)
      if (e != i0 && wt[e] > m1v) { m1v = wt[e]; i1 = e; }

#pragma unroll
    for (int e = 0; e < NEXP; ++e) {
      log_probs_out[(size_t)t * NEXP + e] = lp[e];
      weights_out  [(size_t)t * NEXP + e] = wt[e];
      gw           [(size_t)t * NEXP + e] = (e == i0 || e == i1) ? wt[e] : 0.f;
    }
    top_idx_out[(size_t)t * TOPK + 0] = i0;
    top_idx_out[(size_t)t * TOPK + 1] = i1;
  }
}

// ---------------------------------------------------------------------------
// Fused expert MLP. 256 threads = 8 wave32. 32 tokens per block.
// ---------------------------------------------------------------------------
__global__ __launch_bounds__(256, 1)
void moe_fused(const float* __restrict__ x,
               const float* __restrict__ W1,
               const float* __restrict__ b1,
               const float* __restrict__ W2,
               const float* __restrict__ b2p,
               const float* __restrict__ gw,
               float* __restrict__ out,
               int Ntok) {
  __shared__ unsigned short sA[BM * D_IN];         // x block bf16 (64KB)
  __shared__ float raw1[2][KSLAB * CHUNK];         // TDM W1 fp32 slabs (2x32KB)
  __shared__ unsigned short sB1[16 * 512];         // W1 B-frag bf16 (16KB)
  __shared__ unsigned short sH[4 * 512];           // H chunk A-frag (4KB)
  __shared__ unsigned short sB2[64 * 512];         // W2 B-frag bf16 (64KB)
  __shared__ float sG[NEXP][BM];                   // gates[e][row]

  const int tid  = threadIdx.x;
  const int lane = tid & 31;
  const int w    = tid >> 5;
  const int tokBase = blockIdx.x * BM;

  if (tid < NEXP * BM) {
    int e = tid / BM, r = tid % BM;
    sG[e][r] = gw[(size_t)(tokBase + r) * NEXP + e];
  }

  // ---- stage x -> sA (bf16, row-major) ----
  {
    int row = tid >> 3;
    int colBase = (tid & 7) * 128;
    const float* xp = x + (size_t)(tokBase + row) * D_IN + colBase;
    unsigned int* dst = (unsigned int*)&sA[row * D_IN + colBase];
#pragma unroll 4
    for (int i = 0; i < 32; ++i) {
      float4 v = ((const float4*)xp)[i];
      dst[2 * i]     = ((unsigned)f2bf(v.y) << 16) | f2bf(v.x);
      dst[2 * i + 1] = ((unsigned)f2bf(v.w) << 16) | f2bf(v.z);
    }
  }
  __syncthreads();

  v8f C2[16];
#pragma unroll
  for (int t = 0; t < 16; ++t) C2[t] = (v8f){};

  const int m1 = w & 1, n1 = w >> 1;   // GEMM1 wave tile
  const int m2 = w & 1, j2 = w >> 1;   // GEMM2 wave tile
  const unsigned rawOff0 = (unsigned)(size_t)&raw1[0][0];
  const unsigned rawOff1 = (unsigned)(size_t)&raw1[1][0];

  for (int e = 0; e < NEXP; ++e) {
    const float* W1e = W1 + (size_t)e * D_IN * D_HID;
    const float* W2e = W2 + (size_t)e * D_HID * D_OUT;
    const float* b1e = b1 + (size_t)e * D_HID;

    for (int nb = 0; nb < D_HID; nb += CHUNK) {
      // ============ GEMM1: H[32x64] = x[32x1024] @ W1[:, nb:nb+64] =========
      v8f C1 = (v8f){};
      // prime the TDM pipeline: slab 0 -> raw1[0]
      if (w == 0) tdm_load_w1(W1e + nb, rawOff0);

      for (int s = 0; s < NSLAB; ++s) {
        if (w == 0) {
          if (s + 1 < NSLAB) {
            // overlap next DMA with this slab's compute; TENSORcnt is
            // in-order, so <=1 outstanding means the older DMA has landed.
            tdm_load_w1(W1e + (size_t)(s + 1) * KSLAB * D_HID + nb,
                        ((s + 1) & 1) ? rawOff1 : rawOff0);
            __builtin_amdgcn_s_wait_tensorcnt(1);
          } else {
            __builtin_amdgcn_s_wait_tensorcnt(0);
          }
        }
        __syncthreads();   // TDM data visible to all waves

        // convert raw fp32 slab -> bf16 B-fragment layout (32 elems/thread)
        {
          const float* src = &raw1[s & 1][0];
          int krow = tid >> 1;            // 0..127
          int cb   = (tid & 1) * 32;
          int kc = krow >> 5, kk = krow & 31;
          int l0 = ((kk >> 4) << 4);
          int vg = (kk & 15) >> 1;
          int pr = kk & 1;
          const float4* sp = (const float4*)(src + krow * CHUNK + cb);
          int tb = kc * 4;                // tile row base
#pragma unroll
          for (int i4 = 0; i4 < 8; ++i4) {
            float4 v = sp[i4];
            int n0 = cb + i4 * 4;
            sB1[(tb + ((n0    ) >> 4)) * 512 + (((n0    ) & 15) + l0) * 16 + vg * 2 + pr] = f2bf(v.x);
            sB1[(tb + ((n0 + 1) >> 4)) * 512 + (((n0 + 1) & 15) + l0) * 16 + vg * 2 + pr] = f2bf(v.y);
            sB1[(tb + ((n0 + 2) >> 4)) * 512 + (((n0 + 2) & 15) + l0) * 16 + vg * 2 + pr] = f2bf(v.z);
            sB1[(tb + ((n0 + 3) >> 4)) * 512 + (((n0 + 3) & 15) + l0) * 16 + vg * 2 + pr] = f2bf(v.w);
          }
        }
        __syncthreads();

        // 4 WMMA k-steps over this slab
        {
          int arow = m1 * 16 + (lane & 15);
          int aoff = ((lane >> 4) ? 16 : 0);
#pragma unroll
          for (int kc = 0; kc < 4; ++kc) {
            int ks = s * KSLAB + kc * 32;
            FragBF fa, fb;
            const uint4* ap = (const uint4*)&sA[arow * D_IN + ks + aoff];
            fa.q[0] = ap[0];
            fa.q[1] = ap[1];
            const uint4* bp = (const uint4*)&sB1[(kc * 4 + n1) * 512 + lane * 16];
            fb.q[0] = bp[0];
            fb.q[1] = bp[1];
            C1 = __builtin_amdgcn_wmma_f32_16x16x32_bf16(
                false, fa.v, false, fb.v, (short)0, C1, false, false);
          }
        }
        __syncthreads();   // sB1/raw reuse safe for next stage
      }

      // bias + exact GELU + gate-fold, scatter into A-frag layout for GEMM2
      {
        float bias1 = b1e[nb + n1 * 16 + (lane & 15)];
        int kH = n1 * 16 + (lane & 15);
        int kc = kH >> 5, kk = kH & 31;
        int lh = ((kk >> 4) & 1) * 16;
        int vg = (kk & 15) >> 1;
        int pr = kk & 1;
#pragma unroll
        for (int v = 0; v < 8; ++v) {
          int row = m1 * 16 + v + 8 * (lane >> 4);
          float hv = C1[v] + bias1;
          hv = 0.5f * hv * (1.0f + erff(hv * 0.70710678118f));
          hv *= sG[e][row];
          int mt = row >> 4, rr = row & 15;
          sH[(mt * 2 + kc) * 512 + (rr + lh) * 16 + vg * 2 + pr] = f2bf(hv);
        }
      }
      __syncthreads();

      // ====== GEMM2 partial: out[32x1024] += Hc[32x64] @ W2[nb:nb+64,:] ====
      for (int ks2 = 0; ks2 < 2; ++ks2) {
        {
          int krow = tid >> 3;
          int cb   = (tid & 7) * 128;
          const float* wp2 = W2e + (size_t)(nb + ks2 * 32 + krow) * D_OUT + cb;
          __builtin_prefetch(wp2 + (size_t)32 * D_OUT, 0, 1);
          int l0 = ((krow >> 4) << 4);
          int vg = (krow & 15) >> 1;
          int pr = krow & 1;
#pragma unroll 4
          for (int i4 = 0; i4 < 32; ++i4) {
            float4 v = ((const float4*)wp2)[i4];
            int n0 = cb + i4 * 4;
            sB2[((n0    ) >> 4) * 512 + (((n0    ) & 15) + l0) * 16 + vg * 2 + pr] = f2bf(v.x);
            sB2[((n0 + 1) >> 4) * 512 + (((n0 + 1) & 15) + l0) * 16 + vg * 2 + pr] = f2bf(v.y);
            sB2[((n0 + 2) >> 4) * 512 + (((n0 + 2) & 15) + l0) * 16 + vg * 2 + pr] = f2bf(v.z);
            sB2[((n0 + 3) >> 4) * 512 + (((n0 + 3) & 15) + l0) * 16 + vg * 2 + pr] = f2bf(v.w);
          }
        }
        __syncthreads();
        FragBF fa2;
        {
          const uint4* ap2 = (const uint4*)&sH[(m2 * 2 + ks2) * 512 + lane * 16];
          fa2.q[0] = ap2[0];
          fa2.q[1] = ap2[1];
        }
#pragma unroll
        for (int t = 0; t < 16; ++t) {
          FragBF fb2;
          const uint4* bp2 = (const uint4*)&sB2[(j2 * 16 + t) * 512 + lane * 16];
          fb2.q[0] = bp2[0];
          fb2.q[1] = bp2[1];
          C2[t] = __builtin_amdgcn_wmma_f32_16x16x32_bf16(
              false, fa2.v, false, fb2.v, (short)0, C2[t], false, false);
        }
        __syncthreads();
      }
    } // nb chunks
  } // experts

  // ---- epilogue: out = C2 + sum_e gate[row][e] * b2[e][col] ----
#pragma unroll
  for (int t = 0; t < 16; ++t) {
    int col = j2 * 256 + t * 16 + (lane & 15);
    float bv[NEXP];
#pragma unroll
    for (int e = 0; e < NEXP; ++e) bv[e] = b2p[(size_t)e * D_OUT + col];
#pragma unroll
    for (int v = 0; v < 8; ++v) {
      int row = m2 * 16 + v + 8 * (lane >> 4);
      float bias = 0.f;
#pragma unroll
      for (int e = 0; e < NEXP; ++e) bias += sG[e][row] * bv[e];
      out[(size_t)(tokBase + row) * D_OUT + col] = C2[t][v] + bias;
    }
  }
}

// ---------------------------------------------------------------------------
extern "C" void kernel_launch(void* const* d_in, const int* in_sizes, int n_in,
                              void* d_out, int out_size, void* d_ws, size_t ws_size,
                              hipStream_t stream) {
  const float* x   = (const float*)d_in[0];
  const float* mus = (const float*)d_in[1];
  const float* ls  = (const float*)d_in[2];
  const float* W1  = (const float*)d_in[3];
  const float* b1  = (const float*)d_in[4];
  const float* W2  = (const float*)d_in[5];
  const float* b2  = (const float*)d_in[6];

  int Ntok = in_sizes[0] / D_IN;     // 8192

  float* out_final  = (float*)d_out;
  float* log_probs  = out_final + (size_t)Ntok * D_OUT;
  float* weights    = log_probs + (size_t)Ntok * NEXP;
  int*   top_idx    = (int*)(weights + (size_t)Ntok * NEXP);
  float* gw         = (float*)d_ws;  // [Ntok][NEXP] gated weights

  routing_kernel<<<(Ntok + 7) / 8, 256, 0, stream>>>(
      x, mus, ls, log_probs, weights, top_idx, gw, Ntok);

  moe_fused<<<Ntok / BM, 256, 0, stream>>>(
      x, W1, b1, W2, b2, gw, out_final, Ntok);
}
